// MultiHeadAttentionBlock_34832184771293
// MI455X (gfx1250) — compile-verified, matrix operand order checked by reading, not run on hardware
//
#include <hip/hip_runtime.h>

// ---------------- problem constants ----------------
constexpr int B_      = 8;
constexpr int C_      = 512;
constexpr int N_      = 1024;   // H*W = 32*32
constexpr int HEADS_  = 8;
constexpr int D_      = C_ / HEADS_;   // 64
constexpr int GROUPS_ = 32;
constexpr int CG_     = C_ / GROUPS_;  // 16
constexpr float EPS_  = 1e-5f;

// ---------------- WMMA fragment types ----------------
typedef __attribute__((ext_vector_type(8)))  _Float16 v8h;
typedef __attribute__((ext_vector_type(16))) _Float16 v16h;
typedef __attribute__((ext_vector_type(8)))  float    v8f;
// TDM descriptor vector types (probe-verified builtin signature)
typedef __attribute__((ext_vector_type(4))) unsigned int u32x4;
typedef __attribute__((ext_vector_type(8))) int          i32x8;
typedef __attribute__((ext_vector_type(4))) int          i32x4;

// The TDM builtin carries the LDS target only inside descriptor *values*, so
// the compiler cannot see that it writes the shared-memory tile. A blank
// "memory" clobber is NOT enough: the tile address never escapes, so alias
// analysis proves the asm can't touch it and still folds the ds_loads to
// undef (observed rounds 2-3). Passing the pointer as an asm operand escapes
// the object, making the clobber effective while keeping loads on the
// ds_load path.
__device__ inline void lds_publish(const void* p) {
    asm volatile("" :: "v"(p) : "memory");
}

// 16-bit 16x32 (MxK) fragment layout (CDNA5 ISA 7.12.2):
//   lanes 0-15 : row = lane,    elems 0-7 -> K=0..7,  elems 8-15 -> K=16..23
//   lanes 16-31: row = lane-16, elems 0-7 -> K=8..15, elems 8-15 -> K=24..31
// B fragments (KxN) use the symmetric layout with N in place of M.
__device__ inline const _Float16* lane_frag_ptr(const _Float16* base, int lane, int rowStride) {
    int idx  = lane & 15;
    int koff = (lane >> 4) << 3;   // 0 or 8
    return base + (size_t)idx * rowStride + koff;
}

__device__ inline v16h load_frag(const _Float16* p) {
    v8h lo = *(const v8h*)(p);
    v8h hi = *(const v8h*)(p + 16);
    return __builtin_shufflevector(lo, hi, 0,1,2,3,4,5,6,7,8,9,10,11,12,13,14,15);
}

__device__ inline v8f wmma16(v16h a, v16h b, v8f c) {
    return __builtin_amdgcn_wmma_f32_16x16x32_f16(
        /*neg_a=*/false, a, /*neg_b=*/false, b,
        /*c_mod=*/(short)0, c, /*reuse_a=*/false, /*reuse_b=*/false);
}

// ---------------------------------------------------------------------
// Tensor Data Mover: async 2D tile load (Global -> LDS).
//   Tile: tile_dim0 = 32 f16 (contiguous c), tile_dim1 = rows (n),
//   row stride = C_ elements. Packed row-major into LDS at lds_byte_off.
// Descriptor per CDNA5 ISA 08_async_tensor §8 (D# groups 0/1; 2D => groups
// 2/3 unused). Tracked by TENSORcnt.
// ---------------------------------------------------------------------
__device__ inline void tdm_load_tile(unsigned long long gaddr, unsigned lds_byte_off,
                                     int rows) {
    u32x4 g0;
    g0[0] = 1u;                                            // count=1, user desc
    g0[1] = lds_byte_off;                                  // lds_addr (bytes)
    g0[2] = (unsigned)(gaddr & 0xffffffffull);             // global_addr[31:0]
    g0[3] = ((unsigned)(gaddr >> 32) & 0x01ffffffu)        // global_addr[56:32]
            | 0x80000000u;                                 // type=2 ("image")
    i32x8 g1;
    g1[0] = (int)(1u << 16);                               // data_size=1 -> 2 bytes
    g1[1] = (int)((unsigned)C_ << 16);                     // tensor_dim0 = C_ (lo16)
    g1[2] = (int)((unsigned)N_ << 16);                     // tensor_dim1 = N_ (lo16)
    g1[3] = (int)(32u << 16);                              // tile_dim0 = 32
    g1[4] = rows;                                          // tile_dim1 ; tile_dim2=0
    g1[5] = C_;                                            // tensor_dim0_stride lo32
    g1[6] = 0;
    g1[7] = 0;
    i32x4 z4 = {0, 0, 0, 0};
#if __clang_major__ >= 23
    i32x8 z8 = {0, 0, 0, 0, 0, 0, 0, 0};
    __builtin_amdgcn_tensor_load_to_lds(g0, g1, z4, z4, z8, 0);
#else
    __builtin_amdgcn_tensor_load_to_lds(g0, g1, z4, z4, 0);
#endif
}

// =====================================================================
// Kernel 1: GroupNorm (fp32 stats) -> f16 activations, layout [B, N, C]
// =====================================================================
__global__ void gn_kernel(const float* __restrict__ x,
                          const float* __restrict__ gs,
                          const float* __restrict__ gb,
                          _Float16* __restrict__ hn) {
    int b = blockIdx.x / GROUPS_;
    int g = blockIdx.x % GROUPS_;
    const float* xg = x + ((size_t)b * C_ + (size_t)g * CG_) * N_;

    float s = 0.f, s2 = 0.f;
    for (int i = threadIdx.x; i < CG_ * N_; i += blockDim.x) {
        float v = xg[i];
        s += v; s2 += v * v;
    }
    for (int m = 16; m >= 1; m >>= 1) {
        s  += __shfl_xor(s,  m);
        s2 += __shfl_xor(s2, m);
    }
    __shared__ float red[18];
    int wave = threadIdx.x >> 5, lane = threadIdx.x & 31;
    if (lane == 0) { red[wave] = s; red[8 + wave] = s2; }
    __syncthreads();
    if (threadIdx.x == 0) {
        float ts = 0.f, ts2 = 0.f;
        for (int i = 0; i < 8; ++i) { ts += red[i]; ts2 += red[8 + i]; }
        float inv = 1.0f / (float)(CG_ * N_);
        float mu  = ts * inv;
        float var = ts2 * inv - mu * mu;
        red[16] = mu;
        red[17] = rsqrtf(var + EPS_);
    }
    __syncthreads();
    float mu = red[16], rstd = red[17];

    for (int i = threadIdx.x; i < CG_ * N_; i += blockDim.x) {
        int cg = i / N_;
        int n  = i - cg * N_;
        int c  = g * CG_ + cg;
        float v = (xg[i] - mu) * rstd * gs[c] + gb[c];
        hn[((size_t)b * N_ + n) * C_ + c] = (_Float16)v;
    }
}

// =====================================================================
// Kernel 2: fp32 -> f16 weight conversion
// =====================================================================
__global__ void w2h_kernel(const float* __restrict__ w, _Float16* __restrict__ wh, int count) {
    int i = blockIdx.x * blockDim.x + threadIdx.x;
    if (i < count) wh[i] = (_Float16)w[i];
}

// =====================================================================
// Kernel 3: channel-mixing GEMM with TDM-staged activations.
//   Y[o,n] = sum_c W[o,c] * X[b,n,c] (+ bias, + optional residual)
//   W: f16 [C,C] row-major (A frags, direct global, L2-resident)
//   X: f16 [B,N,C]         (128n x 32c tile DMA'd to LDS per K-step)
// Block = 256 thr = 8 waves = 2(o) x 4(n); block tile 64(o) x 128(n).
// Double-buffered TDM: DMA of step i+1 overlaps WMMA of step i.
//   mode 0: store f16 transposed  Y[(b*N+n)*C + o]   (q, k)
//   mode 1: store f16 normal      Y[(b*C+o)*N + n]   (v)
//   mode 2: store f32 + bias + residual -> out [B,C,N] (final proj)
// =====================================================================
__global__ void gemm_kernel(const _Float16* __restrict__ Wh,
                            const float*    __restrict__ bias,
                            const _Float16* __restrict__ X,
                            _Float16*       __restrict__ Yh,
                            const float*    __restrict__ xres,
                            float*          __restrict__ Yf,
                            int mode) {
    constexpr int OBT = C_ / 64;    // 8 block tiles along o
    constexpr int NBT = N_ / 128;   // 8 block tiles along n
    constexpr unsigned TILE_BYTES = 128 * 32 * 2;   // 8 KB

    int blk = blockIdx.x;
    int b   = blk / (OBT * NBT);
    int t   = blk % (OBT * NBT);
    int blockO = (t / NBT) * 64;
    int blockN = (t % NBT) * 128;
    int wave = threadIdx.x >> 5, lane = threadIdx.x & 31;
    int ot   = wave >> 2;           // 0..1
    int nloc = (wave & 3) * 32;     // 0,32,64,96
    int obase = blockO + ot * 32;

    __shared__ __align__(16) _Float16 xtile[2][128 * 32];

    const _Float16* Xrow = X + ((size_t)b * N_ + blockN) * C_;   // tile origin

    v8f acc[2][2] = {{{}, {}}, {{}, {}}};
    int cur = 0;
    if (wave == 0)
        tdm_load_tile((unsigned long long)(uintptr_t)(Xrow + 0), 0u, 128);

    for (int cb = 0; cb < C_; cb += 32) {
        if (wave == 0) __builtin_amdgcn_s_wait_tensorcnt(0);
        __syncthreads();              // publish LDS tile `cur`; retire `cur^1`
        lds_publish(&xtile[0][0]);    // escape tile + clobber: force ds_loads
        if (wave == 0 && cb + 32 < C_)
            tdm_load_tile((unsigned long long)(uintptr_t)(Xrow + cb + 32),
                          (unsigned)(cur ^ 1) * TILE_BYTES, 128);

        const _Float16* xb = &xtile[cur][0];
        v16h a0 = load_frag(lane_frag_ptr(Wh + (size_t)(obase +  0) * C_ + cb, lane, C_));
        v16h a1 = load_frag(lane_frag_ptr(Wh + (size_t)(obase + 16) * C_ + cb, lane, C_));
        v16h b0 = load_frag(lane_frag_ptr(xb + (size_t)(nloc +  0) * 32, lane, 32));
        v16h b1 = load_frag(lane_frag_ptr(xb + (size_t)(nloc + 16) * 32, lane, 32));
        acc[0][0] = wmma16(a0, b0, acc[0][0]);
        acc[0][1] = wmma16(a0, b1, acc[0][1]);
        acc[1][0] = wmma16(a1, b0, acc[1][0]);
        acc[1][1] = wmma16(a1, b1, acc[1][1]);
        cur ^= 1;
    }

    int idx   = lane & 15;
    int mrow0 = (lane >> 4) << 3;   // 0 or 8
    for (int i = 0; i < 2; ++i)
        for (int j = 0; j < 2; ++j) {
            int n = blockN + nloc + j * 16 + idx;
            for (int r = 0; r < 8; ++r) {
                int o = obase + i * 16 + mrow0 + r;
                float v = acc[i][j][r] + bias[o];
                if (mode == 0) {
                    Yh[((size_t)b * N_ + n) * C_ + o] = (_Float16)v;
                } else if (mode == 1) {
                    Yh[((size_t)b * C_ + o) * N_ + n] = (_Float16)v;
                } else {
                    size_t oidx = ((size_t)b * C_ + o) * N_ + n;
                    Yf[oidx] = v + xres[oidx];
                }
            }
        }
}

// =====================================================================
// Kernel 4: flash attention per (b, head).
//   Q, K: f16 [B,N,C] ; V: f16 [B,C,N] ; OH (out): f16 [B,N,C]
// Block = 128 threads = 4 waves; each wave owns 16 queries (block: 64).
// Streams keys 32 at a time with online softmax (fp32 stats/accum).
// =====================================================================
__global__ void attn_kernel(const _Float16* __restrict__ Q,
                            const _Float16* __restrict__ K,
                            const _Float16* __restrict__ V,
                            _Float16*       __restrict__ OH) {
    constexpr int QT = N_ / 64;
    int wave = threadIdx.x >> 5;
    int lane = threadIdx.x & 31;
    int qt = blockIdx.x % QT;
    int bh = blockIdx.x / QT;
    int h  = bh % HEADS_;
    int b  = bh / HEADS_;
    int qbase = qt * 64 + wave * 16;
    int cbase = h * D_;

    const _Float16* Qb = Q + (size_t)b * N_ * C_;
    const _Float16* Kb = K + (size_t)b * N_ * C_;
    const _Float16* Vb = V + (size_t)b * C_ * N_;

    v16h aq0 = load_frag(lane_frag_ptr(Qb + (size_t)qbase * C_ + cbase +  0, lane, C_));
    v16h aq1 = load_frag(lane_frag_ptr(Qb + (size_t)qbase * C_ + cbase + 32, lane, C_));

    v8f acc[4] = {{}, {}, {}, {}};
    float rmax[8], rsum[8], alpha[8];
    for (int r = 0; r < 8; ++r) { rmax[r] = -3.0e38f; rsum[r] = 0.f; }

    __shared__ __align__(16) _Float16 plds[4][16 * 32];
    _Float16* myP = plds[wave];
    int idx = lane & 15;
    const float scale = 0.125f;          // 1/sqrt(64)

    for (int kb = 0; kb < N_; kb += 32) {
        v8f s[2];
        for (int t = 0; t < 2; ++t) {
            int kpos = kb + t * 16;
            v16h bk0 = load_frag(lane_frag_ptr(Kb + (size_t)kpos * C_ + cbase +  0, lane, C_));
            v16h bk1 = load_frag(lane_frag_ptr(Kb + (size_t)kpos * C_ + cbase + 32, lane, C_));
            v8f c = {};
            c = wmma16(aq0, bk0, c);
            c = wmma16(aq1, bk1, c);
            s[t] = c;
        }
        float bmax[8];
        for (int r = 0; r < 8; ++r) {
            s[0][r] *= scale; s[1][r] *= scale;
            bmax[r] = fmaxf(s[0][r], s[1][r]);
        }
        for (int m = 8; m >= 1; m >>= 1)
            for (int r = 0; r < 8; ++r)
                bmax[r] = fmaxf(bmax[r], __shfl_xor(bmax[r], m, 16));
        float bsum[8];
        for (int r = 0; r < 8; ++r) {
            float mnew = fmaxf(rmax[r], bmax[r]);
            alpha[r] = __expf(rmax[r] - mnew);
            rmax[r] = mnew;
            float p0 = __expf(s[0][r] - mnew);
            float p1 = __expf(s[1][r] - mnew);
            s[0][r] = p0; s[1][r] = p1;
            bsum[r] = p0 + p1;
        }
        for (int m = 8; m >= 1; m >>= 1)
            for (int r = 0; r < 8; ++r)
                bsum[r] += __shfl_xor(bsum[r], m, 16);
        for (int r = 0; r < 8; ++r)
            rsum[r] = rsum[r] * alpha[r] + bsum[r];

        // stage P (16q x 32k f16) through LDS to reshape into an A fragment
        int mrow0 = (lane >> 4) << 3;
        for (int t = 0; t < 2; ++t)
            for (int r = 0; r < 8; ++r)
                myP[(mrow0 + r) * 32 + t * 16 + idx] = (_Float16)s[t][r];
        __syncthreads();

        v16h ap = load_frag(lane_frag_ptr(myP, lane, 32));

        for (int j = 0; j < 4; ++j) {
            v16h bv = load_frag(lane_frag_ptr(
                Vb + (size_t)(cbase + j * 16) * N_ + kb, lane, N_));
            v8f a = acc[j];
            for (int r = 0; r < 8; ++r) a[r] *= alpha[r];
            acc[j] = wmma16(ap, bv, a);
        }
        __syncthreads();
    }

    int mrow0 = (lane >> 4) << 3;
    for (int j = 0; j < 4; ++j)
        for (int r = 0; r < 8; ++r) {
            int m = mrow0 + r;
            float v = acc[j][r] / rsum[r];
            OH[((size_t)b * N_ + qbase + m) * C_ + cbase + j * 16 + idx] = (_Float16)v;
        }
}

// =====================================================================
// Host launcher
// =====================================================================
extern "C" void kernel_launch(void* const* d_in, const int* in_sizes, int n_in,
                              void* d_out, int out_size, void* d_ws, size_t ws_size,
                              hipStream_t stream) {
    (void)in_sizes; (void)n_in; (void)out_size; (void)ws_size;
    const float* x  = (const float*)d_in[0];
    const float* gs = (const float*)d_in[1];
    const float* gb = (const float*)d_in[2];
    const float* wq = (const float*)d_in[3];
    const float* bq = (const float*)d_in[4];
    const float* wk = (const float*)d_in[5];
    const float* bk = (const float*)d_in[6];
    const float* wv = (const float*)d_in[7];
    const float* bv = (const float*)d_in[8];
    const float* wo = (const float*)d_in[9];
    const float* bo = (const float*)d_in[10];
    float* out = (float*)d_out;

    _Float16* ws = (_Float16*)d_ws;
    size_t off = 0;
    const size_t act = (size_t)B_ * N_ * C_;
    const size_t wsz = (size_t)C_ * C_;
    _Float16* hn  = ws + off; off += act;
    _Float16* wqh = ws + off; off += wsz;
    _Float16* wkh = ws + off; off += wsz;
    _Float16* wvh = ws + off; off += wsz;
    _Float16* woh = ws + off; off += wsz;
    _Float16* qh  = ws + off; off += act;      // [B,N,C]
    _Float16* kh  = ws + off; off += act;      // [B,N,C]
    _Float16* vh  = ws + off; off += act;      // [B,C,N]
    _Float16* oh  = ws + off; off += act;      // [B,N,C]

    gn_kernel<<<B_ * GROUPS_, 256, 0, stream>>>(x, gs, gb, hn);

    const int wcnt = C_ * C_;
    w2h_kernel<<<wcnt / 256, 256, 0, stream>>>(wq, wqh, wcnt);
    w2h_kernel<<<wcnt / 256, 256, 0, stream>>>(wk, wkh, wcnt);
    w2h_kernel<<<wcnt / 256, 256, 0, stream>>>(wv, wvh, wcnt);
    w2h_kernel<<<wcnt / 256, 256, 0, stream>>>(wo, woh, wcnt);

    const int gemmBlocks = B_ * (C_ / 64) * (N_ / 128);   // 512
    gemm_kernel<<<gemmBlocks, 256, 0, stream>>>(wqh, bq, hn, qh, nullptr, nullptr, 0);
    gemm_kernel<<<gemmBlocks, 256, 0, stream>>>(wkh, bk, hn, kh, nullptr, nullptr, 0);
    gemm_kernel<<<gemmBlocks, 256, 0, stream>>>(wvh, bv, hn, vh, nullptr, nullptr, 1);

    attn_kernel<<<B_ * HEADS_ * (N_ / 64), 128, 0, stream>>>(qh, kh, vh, oh);

    gemm_kernel<<<gemmBlocks, 256, 0, stream>>>(woh, bo, oh, nullptr, x, out, 2);
}
